// Quanv3x3LayerClass_61753039782313
// MI455X (gfx1250) — matrix operand order repaired and checked by reading, not compiled, not run on hardware
//
#include <hip/hip_runtime.h>

typedef __attribute__((ext_vector_type(16))) _Float16 v16h;
typedef __attribute__((ext_vector_type(8)))  _Float16 v8h;
typedef __attribute__((ext_vector_type(8)))  float    v8f;

#define NPATCH 3136   // 16*14*14 patches
#define KC     12     // 4 kernels * 3 channels
#define DIM    512    // 2^9 amplitudes

// ---------------------------------------------------------------------------
// Stage 1: build Wt[kc][n][k] = U_{kc}[n,k] as f16 (B-operand-friendly layout).
// One wave simulates U e_k for basis state k: 512 amps = 16 f32 regs/lane.
// Amplitude index bits [8:4] -> lane, bits [3:0] -> register.
// Qubit q corresponds to amplitude bit (8-q).
// ---------------------------------------------------------------------------

__device__ __forceinline__ void ry_lane_m(float v[16], float c, float ss, int lm) {
#pragma unroll
  for (int r = 0; r < 16; ++r) {
    float p = __shfl_xor(v[r], lm, 32);
    v[r] = c * v[r] + ss * p;   // ss = +s if this lane holds the |1> half, else -s
  }
}

template <int RM>
__device__ __forceinline__ void ry_reg_m(float v[16], float c, float s) {
#pragma unroll
  for (int r = 0; r < 16; ++r) {
    if (!(r & RM)) {
      float A = v[r], B = v[r | RM];
      v[r]      = c * A - s * B;
      v[r | RM] = s * A + c * B;
    }
  }
}

// CNOT: both ctrl and tgt bits live in the lane dimension
__device__ __forceinline__ void cnot_ll(float v[16], int cm, int tm, int lane) {
  int src = (lane & cm) ? (lane ^ tm) : lane;
#pragma unroll
  for (int r = 0; r < 16; ++r) v[r] = __shfl(v[r], src, 32);
}

// CNOT: ctrl bit in lane dim, tgt bit in register dim (cndmask swap)
template <int TM>
__device__ __forceinline__ void cnot_lr(float v[16], int cm, int lane) {
  bool f = (lane & cm) != 0;
#pragma unroll
  for (int r = 0; r < 16; ++r) {
    if (!(r & TM)) {
      float A = v[r], B = v[r | TM];
      v[r]      = f ? B : A;
      v[r | TM] = f ? A : B;
    }
  }
}

// CNOT: both bits in register dim -> pure register renaming (free)
template <int CM, int TM>
__device__ __forceinline__ void cnot_rr(float v[16]) {
#pragma unroll
  for (int r = 0; r < 16; ++r) {
    if ((r & CM) && !(r & TM)) {
      float t = v[r]; v[r] = v[r | TM]; v[r | TM] = t;
    }
  }
}

// CNOT: ctrl bit in register dim, tgt bit in lane dim
template <int CM>
__device__ __forceinline__ void cnot_rl(float v[16], int ltm) {
#pragma unroll
  for (int r = 0; r < 16; ++r)
    if (r & CM) v[r] = __shfl_xor(v[r], ltm, 32);
}

__global__ __launch_bounds__(256) void build_u_kernel(const float* __restrict__ qp,
                                                      _Float16* __restrict__ Wt) {
  const int lane  = threadIdx.x & 31;
  const int gid   = blockIdx.x * 8 + (threadIdx.x >> 5);  // 0..6143
  const int kc    = gid >> 9;
  const int basis = gid & 511;

  const float* th = qp + kc * 16;
  float c[16], s[16];
#pragma unroll
  for (int t = 0; t < 16; ++t) {
    float sv, cv;
    sincosf(0.5f * th[t], &sv, &cv);
    s[t] = sv; c[t] = cv;
  }

  float v[16];
#pragma unroll
  for (int r = 0; r < 16; ++r) v[r] = (lane * 16 + r == basis) ? 1.0f : 0.0f;

  // trainable RY layer 1 (qubits 0..8)
#pragma unroll
  for (int q = 0; q < 5; ++q) {                  // qubits 0..4 -> lane bits 4..0
    int lm = 1 << (4 - q);
    float ss = (lane & lm) ? s[q] : -s[q];
    ry_lane_m(v, c[q], ss, lm);
  }
  ry_reg_m<8>(v, c[5], s[5]);                    // qubit 5 -> reg bit 3
  ry_reg_m<4>(v, c[6], s[6]);
  ry_reg_m<2>(v, c[7], s[7]);
  ry_reg_m<1>(v, c[8], s[8]);                    // qubit 8 -> reg bit 0

  // CNOT ring (q, q+1 mod 9)
  cnot_ll(v, 16, 8, lane);   // (0,1)
  cnot_ll(v,  8, 4, lane);   // (1,2)
  cnot_ll(v,  4, 2, lane);   // (2,3)
  cnot_ll(v,  2, 1, lane);   // (3,4)
  cnot_lr<8>(v, 1, lane);    // (4,5): ctrl lane-bit0, tgt reg-bit3
  cnot_rr<8, 4>(v);          // (5,6)
  cnot_rr<4, 2>(v);          // (6,7)
  cnot_rr<2, 1>(v);          // (7,8)
  cnot_rl<1>(v, 16);         // (8,0): ctrl reg-bit0, tgt lane-bit4

  // trainable RY layer 2 (qubits 0..6, thetas 9..15)
#pragma unroll
  for (int q = 0; q < 5; ++q) {
    int lm = 1 << (4 - q);
    float ss = (lane & lm) ? s[9 + q] : -s[9 + q];
    ry_lane_m(v, c[9 + q], ss, lm);
  }
  ry_reg_m<8>(v, c[14], s[14]);                  // qubit 5
  ry_reg_m<4>(v, c[15], s[15]);                  // qubit 6

  // store column: Wt[kc][n = lane*16+r][k = basis]
  _Float16* base = Wt + ((size_t)kc * DIM + lane * 16) * DIM + basis;
#pragma unroll
  for (int r = 0; r < 16; ++r) base[(size_t)r * DIM] = (_Float16)v[r];
}

// ---------------------------------------------------------------------------
// Stage 2: product state psi for each (channel, patch) -> f16 row of length 512
// psi[idx] = prod_q ( bit_(8-q)(idx) ? sin(pi*x_q/2) : cos(pi*x_q/2) )
// ---------------------------------------------------------------------------
__global__ __launch_bounds__(256) void build_psi_kernel(const float* __restrict__ x,
                                                        _Float16* __restrict__ Psi) {
  const int lane = threadIdx.x & 31;
  const int gid  = blockIdx.x * 8 + (threadIdx.x >> 5);   // 0..9407
  const int ch   = gid / NPATCH;
  const int p    = gid % NPATCH;
  const int b = p / 196, rem = p % 196, hh = rem / 14, ww = rem % 14;

  float cq[9], sq[9];
#pragma unroll
  for (int i = 0; i < 3; ++i) {
#pragma unroll
    for (int j = 0; j < 3; ++j) {
      int q = i * 3 + j;
      int yy = hh + i - 1, xx = ww + j - 1;
      float val = 0.0f;
      if (yy >= 0 && yy < 14 && xx >= 0 && xx < 14)
        val = x[(((size_t)b * 14 + yy) * 14 + xx) * 3 + ch];
      float sv, cv;
      sincosf(1.5707963267948966f * val, &sv, &cv);   // (pi*x)/2
      cq[q] = cv; sq[q] = sv;
    }
  }

  // lane part: qubits 0..4 <-> lane bits 4..0
  float lp = 1.0f;
#pragma unroll
  for (int q = 0; q < 5; ++q) lp *= ((lane >> (4 - q)) & 1) ? sq[q] : cq[q];

  v8h lo8, hi8;
#pragma unroll
  for (int r = 0; r < 16; ++r) {
    float a = lp;
    a *= (r & 8) ? sq[5] : cq[5];
    a *= (r & 4) ? sq[6] : cq[6];
    a *= (r & 2) ? sq[7] : cq[7];
    a *= (r & 1) ? sq[8] : cq[8];
    if (r < 8) lo8[r] = (_Float16)a; else hi8[r - 8] = (_Float16)a;
  }
  _Float16* base = Psi + ((size_t)ch * NPATCH + p) * DIM + lane * 16;
  *(v8h*)(base)     = lo8;
  *(v8h*)(base + 8) = hi8;
}

// ---------------------------------------------------------------------------
// Stage 3: Phi = Psi_c * U^T via WMMA f32<-f16, fused with
// out = sum_j sign(j)*Phi^2.
// Workgroup tile: 64(M) x 512(N); 8 waves, each 64(M) x 64(N) via a 4x4
// grid of 16x16 accumulators. K-loop over 512 in 32-steps:
// per iter 8 A-loads + 8 B-loads (b128) feed 16 WMMAs (B reused across the
// 4 M-subtiles, A reused across the 4 N-subtiles) -> 4x less L2 traffic
// than a 16-row tile.
// Grid: (49 M-tiles, 12 kc).
// ---------------------------------------------------------------------------
__global__ __launch_bounds__(256) void quanv_gemm_kernel(const _Float16* __restrict__ Psi,
                                                         const _Float16* __restrict__ Wt,
                                                         float* __restrict__ out) {
  __shared__ float part[8][64];
  const int lane  = threadIdx.x & 31;
  const int wv    = threadIdx.x >> 5;
  const int kc    = blockIdx.y;
  const int ch    = kc % 3;
  const int mbase = blockIdx.x * 64;
  const int mlo   = lane & 15;
  const int hi    = lane >> 4;

  // A fragment sources: 4 M-subtiles; row m = ms*16 + mlo,
  // lane-half picks contiguous K 16*hi..16*hi+15 (same order as B).
  const _Float16* Arow[4];
#pragma unroll
  for (int ms = 0; ms < 4; ++ms)
    Arow[ms] = Psi + ((size_t)ch * NPATCH + mbase + ms * 16 + mlo) * DIM;

  // B fragment sources: 4 N-subtiles; Wt rows are n-major, K contiguous.
  const _Float16* Bbase = Wt + (size_t)kc * DIM * DIM;
  const int n0 = wv * 64 + mlo;
  const _Float16* Brow[4];
#pragma unroll
  for (int ns = 0; ns < 4; ++ns)
    Brow[ns] = Bbase + (size_t)(n0 + ns * 16) * DIM + hi * 16;

  v8f acc[4][4];
#pragma unroll
  for (int ms = 0; ms < 4; ++ms)
#pragma unroll
    for (int ns = 0; ns < 4; ++ns) acc[ms][ns] = (v8f){};

  for (int k0 = 0; k0 < DIM; k0 += 32) {
    v16h a[4];
#pragma unroll
    for (int ms = 0; ms < 4; ++ms) {
      v8h alo = *(const v8h*)(Arow[ms] + k0 + hi * 16);      // K = k0 + hi*16 + 0..7
      v8h ahi = *(const v8h*)(Arow[ms] + k0 + hi * 16 + 8);  // K = k0 + hi*16 + 8..15
#pragma unroll
      for (int t = 0; t < 8; ++t) { a[ms][t] = alo[t]; a[ms][t + 8] = ahi[t]; }
    }
#pragma unroll
    for (int ns = 0; ns < 4; ++ns) {
      v16h b = *(const v16h*)(Brow[ns] + k0);
#pragma unroll
      for (int ms = 0; ms < 4; ++ms)
        acc[ms][ns] = __builtin_amdgcn_wmma_f32_16x16x32_f16(
            false, a[ms], false, b, (short)0, acc[ms][ns], false, false);
    }
  }

  // C/D layout: col n = lane&15, row m = r + 8*(lane>=16).
  // Square, sum this wave's 64 columns per row (reduce 16-lane column group).
#pragma unroll
  for (int ms = 0; ms < 4; ++ms) {
#pragma unroll
    for (int r = 0; r < 8; ++r) {
      float t = acc[ms][0][r] * acc[ms][0][r] + acc[ms][1][r] * acc[ms][1][r]
              + acc[ms][2][r] * acc[ms][2][r] + acc[ms][3][r] * acc[ms][3][r];
      t += __shfl_xor(t, 1, 32);
      t += __shfl_xor(t, 2, 32);
      t += __shfl_xor(t, 4, 32);
      t += __shfl_xor(t, 8, 32);
      if (mlo == 0) part[wv][ms * 16 + hi * 8 + r] = t;
    }
  }
  __syncthreads();

  if (threadIdx.x < 64) {
    float f = 0.0f;  // waves 0..3 cover n<256 (qubit0=0 -> +), 4..7 -> -
#pragma unroll
    for (int w8 = 0; w8 < 8; ++w8)
      f += (w8 < 4 ? 1.0f : -1.0f) * part[w8][threadIdx.x];
    out[((size_t)(mbase + threadIdx.x)) * 12 + kc] = f;
  }
}

// ---------------------------------------------------------------------------
extern "C" void kernel_launch(void* const* d_in, const int* in_sizes, int n_in,
                              void* d_out, int out_size, void* d_ws, size_t ws_size,
                              hipStream_t stream) {
  const float* x  = (const float*)d_in[0];   // (16,14,14,3) f32
  const float* qp = (const float*)d_in[1];   // (4,3,16) f32
  float* out = (float*)d_out;                // (16,14,14,12) f32

  _Float16* Wt  = (_Float16*)d_ws;                                   // 12*512*512 f16 = 6 MB
  _Float16* Psi = (_Float16*)((char*)d_ws +
                              (size_t)KC * DIM * DIM * sizeof(_Float16)); // 3*3136*512 f16 = 9.2 MB

  // 12*512 basis-state circuit sims, 8 waves/block
  build_u_kernel<<<dim3(768), dim3(256), 0, stream>>>(qp, Wt);
  // 3*3136 product states, 8 waves/block
  build_psi_kernel<<<dim3(1176), dim3(256), 0, stream>>>(x, Psi);
  // 12 GEMMs [3136x512]x[512x512] fused with signed squared reduction
  quanv_gemm_kernel<<<dim3(49, 12), dim3(256), 0, stream>>>(Psi, Wt, out);
}
